// AttentionBlockManual_60129542447
// MI455X (gfx1250) — matrix operand order repaired and checked by reading, not compiled, not run on hardware
//
#include <hip/hip_runtime.h>
#include <hip/hip_bf16.h>

// ---------------------------------------------------------------------------
// AttentionBlock on gfx1250 (CDNA5, wave32, WMMA bf16 16x16x32, f32 accum)
// - vectorized fragment loads (b128 pairs)
// - transposed bf16 weights so B-fragments are contiguous
// - async global->LDS staging of V chunks (ASYNCcnt) in the flash-attention loop
// ---------------------------------------------------------------------------

typedef __attribute__((ext_vector_type(16))) __bf16 v16bf;
typedef __attribute__((ext_vector_type(8)))  float  v8f;

#define N_BATCH 8
#define C_IN    256
#define HS      32
#define WS      32
#define P_TOK   1024          // HS*WS tokens per image
#define T_DIM   512
#define D_KQ    512
#define D_V     64
#define H_HEADS 8
#define DH      64            // D_KQ / H_HEADS
#define D_CAT   512           // H_HEADS * D_V

__device__ __forceinline__ unsigned short f2bf(float f) {
    unsigned u = __builtin_bit_cast(unsigned, f);
    unsigned r = (u + 0x7FFFu + ((u >> 16) & 1u)) >> 16;
    return (unsigned short)r;
}
__device__ __forceinline__ __bf16 bfbits(unsigned short u) {
    return __builtin_bit_cast(__bf16, u);
}
// two contiguous 16-byte runs -> one 16x32 bf16 fragment (per-lane view)
__device__ __forceinline__ v16bf load_frag_pair(const unsigned short* p0,
                                                const unsigned short* p1) {
    struct alignas(32) U2 { uint4 a, b; } u;
    u.a = *(const uint4*)p0;
    u.b = *(const uint4*)p1;
    return __builtin_bit_cast(v16bf, u);
}

// ---------------------------------------------------------------------------
// 1) temb = relu(t @ Wt + bt)
// ---------------------------------------------------------------------------
__global__ __launch_bounds__(256) void temb_kernel(const float* __restrict__ t,
                                                   const float* __restrict__ Wt,
                                                   const float* __restrict__ bt,
                                                   float* __restrict__ temb) {
    int n = blockIdx.x, c = threadIdx.x;
    float s = bt[c];
    for (int i = 0; i < T_DIM; ++i) s = fmaf(t[n * T_DIM + i], Wt[i * C_IN + c], s);
    temb[n * C_IN + c] = fmaxf(s, 0.f);
}

// ---------------------------------------------------------------------------
// 2) BatchNorm statistics per channel over (N,H,W) of xb = x + temb
// ---------------------------------------------------------------------------
__global__ __launch_bounds__(256) void bn_stats_kernel(const float* __restrict__ x,
                                                       const float* __restrict__ temb,
                                                       float* __restrict__ mean,
                                                       float* __restrict__ rstd) {
    int c = blockIdx.x, tid = threadIdx.x;
    float s = 0.f, s2 = 0.f;
    for (int i = tid; i < N_BATCH * P_TOK; i += 256) {
        int n = i >> 10, sp = i & 1023;
        float v = x[(n * C_IN + c) * P_TOK + sp] + temb[n * C_IN + c];
        s += v; s2 += v * v;
    }
    __shared__ float rs[256], rs2[256];
    rs[tid] = s; rs2[tid] = s2;
    __syncthreads();
    for (int st = 128; st > 0; st >>= 1) {
        if (tid < st) { rs[tid] += rs[tid + st]; rs2[tid] += rs2[tid + st]; }
        __syncthreads();
    }
    if (tid == 0) {
        float m = rs[0] * (1.f / 8192.f);
        float var = rs2[0] * (1.f / 8192.f) - m * m;
        mean[c] = m;
        rstd[c] = rsqrtf(var + 1e-5f);
    }
}

// ---------------------------------------------------------------------------
// 3) normalize + transpose(0,3,2,1) + pack bf16:  X[n][p][c], p = w*32 + h
// ---------------------------------------------------------------------------
__global__ __launch_bounds__(256) void norm_kernel(const float* __restrict__ x,
                                                   const float* __restrict__ temb,
                                                   const float* __restrict__ mean,
                                                   const float* __restrict__ rstd,
                                                   const float* __restrict__ gamma,
                                                   const float* __restrict__ beta,
                                                   unsigned short* __restrict__ Xbf) {
    int np = blockIdx.x;          // n*1024 + p
    int n = np >> 10, p = np & 1023;
    int c = threadIdx.x;
    int w = p >> 5, h = p & 31;
    int sp = h * WS + w;          // index into x's [H][W]
    float v = (x[(n * C_IN + c) * P_TOK + sp] + temb[n * C_IN + c] - mean[c]) * rstd[c];
    v = v * gamma[c] + beta[c];
    Xbf[(size_t)np * C_IN + c] = f2bf(v);
}

// ---------------------------------------------------------------------------
// weight conversion with transpose: src[K][N] f32 -> dst[N][K] bf16
// ---------------------------------------------------------------------------
__global__ __launch_bounds__(256) void cvtT_kernel(const float* __restrict__ src,
                                                   unsigned short* __restrict__ dst,
                                                   int Kd, int N) {
    int i = blockIdx.x * 256 + threadIdx.x;  // over Kd*N outputs (dst row-major [N][Kd])
    if (i < Kd * N) {
        int nn = i / Kd, k = i - nn * Kd;
        dst[i] = f2bf(src[k * N + nn]);
    }
}

// ---------------------------------------------------------------------------
// 4) Generic bf16 GEMM: 4 waves/block, one 16x16 C tile per wave.
//    C = A[M,Kd] @ B (B given TRANSPOSED: BT[N][Kd]), bf16 out.
// ---------------------------------------------------------------------------
__global__ __launch_bounds__(128) void gemm_bf16_kernel(const unsigned short* __restrict__ A,
                                                        const unsigned short* __restrict__ BT,
                                                        unsigned short* __restrict__ C,
                                                        int Kd, int N, long strideA, long strideC) {
    int wv = threadIdx.x >> 5;
    int lane = threadIdx.x & 31, g = lane >> 4, mr = lane & 15;
    int n0 = (blockIdx.x * 4 + wv) * 16, m0 = blockIdx.y * 16, b = blockIdx.z;
    A += (long)b * strideA;
    C += (long)b * strideC;
    const unsigned short* arow = A + (long)(m0 + mr) * Kd;
    const unsigned short* brow = BT + (long)(n0 + mr) * Kd;
    v8f acc = {};
    for (int kc = 0; kc < Kd; kc += 32) {
        v16bf a  = load_frag_pair(arow + kc + 8 * g, arow + kc + 16 + 8 * g);
        v16bf bm = load_frag_pair(brow + kc + 8 * g, brow + kc + 16 + 8 * g);
        acc = __builtin_amdgcn_wmma_f32_16x16x32_bf16(false, a, false, bm, (short)0, acc,
                                                      false, false);
    }
#pragma unroll
    for (int r = 0; r < 8; ++r)
        C[(long)(m0 + r + 8 * g) * N + n0 + mr] = f2bf(acc[r]);
}

// ---------------------------------------------------------------------------
// 5) Flash attention: one wave per (n, head, 16-query tile).
//    V chunk staged LDS via async global->LDS (ASYNCcnt); online softmax.
//    Output layout interleaved: O[n][p][x*8 + head]
// ---------------------------------------------------------------------------
__global__ __launch_bounds__(32) void attn_kernel(const unsigned short* __restrict__ Q,
                                                  const unsigned short* __restrict__ K,
                                                  const unsigned short* __restrict__ V,
                                                  unsigned short* __restrict__ O) {
    const int n = blockIdx.z, h = blockIdx.y, q0 = blockIdx.x * 16;
    const int lane = threadIdx.x, g = lane >> 4, mr = lane & 15;
    const float scale = 0.04419417382415922f;  // 1/sqrt(512)

    __shared__ float          s_s[16][32];      // staged scores
    __shared__ unsigned short s_p[16][32];      // staged P (bf16)
    __shared__ unsigned short s_v[32][64];      // staged V chunk (bf16)
    __shared__ float s_m[16], s_l[16], s_alpha[16], s_linv[16];

    // Q A-fragments (K-dim = dh = 64 -> 2 chunks of 32)
    const unsigned short* qrow = Q + ((long)(n * P_TOK + q0 + mr)) * D_KQ + h * DH;
    v16bf aq0 = load_frag_pair(qrow + 8 * g, qrow + 16 + 8 * g);
    v16bf aq1 = load_frag_pair(qrow + 32 + 8 * g, qrow + 48 + 8 * g);

    v8f acc0 = {}, acc1 = {}, acc2 = {}, acc3 = {};
    if (lane < 16) { s_m[lane] = -1e30f; s_l[lane] = 0.f; }
    // per-lane LDS byte address of this lane's V staging row (wave-relative, low 32 bits)
    unsigned ldsoff = (unsigned)(unsigned long long)(const void*)&s_v[lane][0];
    __syncthreads();

    for (int kc = 0; kc < 32; ++kc) {       // 32 chunks x 32 keys = 1024 keys
        const int key0 = kc * 32;
        // async stage V chunk: each lane streams its 128B row via 8x b128
        {
            unsigned long long gp =
                (unsigned long long)(const void*)(V + ((long)(n * P_TOK + key0 + lane)) * D_V);
#define V_ASYNC(OFF)                                                                        \
    asm volatile("global_load_async_to_lds_b128 %0, %1, off offset:" OFF                    \
                 :: "v"(ldsoff), "v"(gp) : "memory")
            V_ASYNC("0");  V_ASYNC("16"); V_ASYNC("32"); V_ASYNC("48");
            V_ASYNC("64"); V_ASYNC("80"); V_ASYNC("96"); V_ASYNC("112");
#undef V_ASYNC
        }
        // S = Q K^T, two 16-key subtiles
#pragma unroll
        for (int sub = 0; sub < 2; ++sub) {
            const int kt = key0 + sub * 16;
            const unsigned short* krow =
                K + ((long)(n * P_TOK + kt + mr)) * D_KQ + h * DH;
            v16bf bk0 = load_frag_pair(krow + 8 * g, krow + 16 + 8 * g);
            v16bf bk1 = load_frag_pair(krow + 32 + 8 * g, krow + 48 + 8 * g);
            v8f s = {};
            s = __builtin_amdgcn_wmma_f32_16x16x32_bf16(false, aq0, false, bk0, (short)0, s,
                                                        false, false);
            s = __builtin_amdgcn_wmma_f32_16x16x32_bf16(false, aq1, false, bk1, (short)0, s,
                                                        false, false);
#pragma unroll
            for (int r = 0; r < 8; ++r) s_s[r + 8 * g][sub * 16 + mr] = s[r] * scale;
        }
        __syncthreads();
        // online softmax row update (row owner = lane < 16)
        if (lane < 16) {
            const int row = lane;
            float rm = -1e30f;
            for (int j = 0; j < 32; ++j) rm = fmaxf(rm, s_s[row][j]);
            float mold = s_m[row];
            float mnew = fmaxf(mold, rm);
            float alpha = __expf(mold - mnew);
            float psum = 0.f;
            for (int j = 0; j < 32; ++j) {
                float p = __expf(s_s[row][j] - mnew);
                s_p[row][j] = f2bf(p);
                psum += p;
            }
            s_l[row] = s_l[row] * alpha + psum;
            s_m[row] = mnew;
            s_alpha[row] = alpha;
        }
        __syncthreads();
        // rescale accumulators
#pragma unroll
        for (int r = 0; r < 8; ++r) {
            float al = s_alpha[r + 8 * g];
            acc0[r] *= al; acc1[r] *= al; acc2[r] *= al; acc3[r] *= al;
        }
        // A fragment from P (16 x 32 keys) - two b128 LDS reads
        v16bf ap = load_frag_pair(&s_p[mr][16 * g], &s_p[mr][16 * g + 8]);
        // ensure async V staging landed before reading s_v
        asm volatile("s_wait_asynccnt 0" ::: "memory");
        // O += P @ V : N = 64 -> 4 column tiles
        v16bf bv;
#pragma unroll
        for (int e = 0; e < 16; ++e) bv[e] = bfbits(s_v[16 * g + e][mr]);
        acc0 = __builtin_amdgcn_wmma_f32_16x16x32_bf16(false, ap, false, bv, (short)0, acc0,
                                                       false, false);
#pragma unroll
        for (int e = 0; e < 16; ++e) bv[e] = bfbits(s_v[16 * g + e][16 + mr]);
        acc1 = __builtin_amdgcn_wmma_f32_16x16x32_bf16(false, ap, false, bv, (short)0, acc1,
                                                       false, false);
#pragma unroll
        for (int e = 0; e < 16; ++e) bv[e] = bfbits(s_v[16 * g + e][32 + mr]);
        acc2 = __builtin_amdgcn_wmma_f32_16x16x32_bf16(false, ap, false, bv, (short)0, acc2,
                                                       false, false);
#pragma unroll
        for (int e = 0; e < 16; ++e) bv[e] = bfbits(s_v[16 * g + e][48 + mr]);
        acc3 = __builtin_amdgcn_wmma_f32_16x16x32_bf16(false, ap, false, bv, (short)0, acc3,
                                                       false, false);
        __syncthreads();
    }
    if (lane < 16) s_linv[lane] = 1.f / s_l[lane];
    __syncthreads();
#pragma unroll
    for (int r = 0; r < 8; ++r) {
        const int row = q0 + r + 8 * g;
        const float li = s_linv[r + 8 * g];
        long base = ((long)(n * P_TOK + row)) * D_CAT + h;   // col = x*8 + head
        O[base + (long)(0 * 16 + mr) * 8] = f2bf(acc0[r] * li);
        O[base + (long)(1 * 16 + mr) * 8] = f2bf(acc1[r] * li);
        O[base + (long)(2 * 16 + mr) * 8] = f2bf(acc2[r] * li);
        O[base + (long)(3 * 16 + mr) * 8] = f2bf(acc3[r] * li);
    }
}

// ---------------------------------------------------------------------------
// 6) Projection GEMM + bias + transpose back to NCHW + residual (fp32 out)
//    B given transposed: WpT[256][512]. 4 waves per block.
// ---------------------------------------------------------------------------
__global__ __launch_bounds__(128) void proj_kernel(const unsigned short* __restrict__ A,
                                                   const unsigned short* __restrict__ BwT,
                                                   const float* __restrict__ bp,
                                                   const float* __restrict__ xres,
                                                   float* __restrict__ out) {
    int wv = threadIdx.x >> 5;
    int lane = threadIdx.x & 31, g = lane >> 4, mr = lane & 15;
    int n0 = (blockIdx.x * 4 + wv) * 16, m0 = blockIdx.y * 16, b = blockIdx.z;
    const unsigned short* arow = A + (long)b * P_TOK * D_CAT + (long)(m0 + mr) * D_CAT;
    const unsigned short* brow = BwT + (long)(n0 + mr) * D_CAT;
    v8f acc = {};
    for (int kc = 0; kc < D_CAT; kc += 32) {
        v16bf a  = load_frag_pair(arow + kc + 8 * g, arow + kc + 16 + 8 * g);
        v16bf bm = load_frag_pair(brow + kc + 8 * g, brow + kc + 16 + 8 * g);
        acc = __builtin_amdgcn_wmma_f32_16x16x32_bf16(false, a, false, bm, (short)0, acc,
                                                      false, false);
    }
#pragma unroll
    for (int r = 0; r < 8; ++r) {
        int p = m0 + r + 8 * g;
        int c = n0 + mr;
        int w = p >> 5, hh = p & 31;
        long oidx = ((long)(b * C_IN + c) * HS + hh) * WS + w;
        out[oidx] = acc[r] + bp[c] + xres[oidx];
    }
}

// ---------------------------------------------------------------------------
// Host side
// ---------------------------------------------------------------------------
extern "C" void kernel_launch(void* const* d_in, const int* in_sizes, int n_in,
                              void* d_out, int out_size, void* d_ws, size_t ws_size,
                              hipStream_t stream) {
    const float* x     = (const float*)d_in[0];
    const float* t     = (const float*)d_in[1];
    const float* Wt    = (const float*)d_in[2];
    const float* bt    = (const float*)d_in[3];
    const float* Wq    = (const float*)d_in[4];
    const float* Wk    = (const float*)d_in[5];
    const float* Wv    = (const float*)d_in[6];
    const float* Wp    = (const float*)d_in[7];
    const float* bp    = (const float*)d_in[8];
    const float* gamma = (const float*)d_in[9];
    const float* beta  = (const float*)d_in[10];
    float* out = (float*)d_out;

    char* ws = (char*)d_ws;
    size_t off = 0;
    auto carve = [&](size_t bytes) { char* p = ws + off; off += (bytes + 255) & ~(size_t)255; return p; };
    float*          temb  = (float*)carve(N_BATCH * C_IN * 4);
    float*          mean  = (float*)carve(C_IN * 4);
    float*          rstd  = (float*)carve(C_IN * 4);
    unsigned short* Xbf   = (unsigned short*)carve((size_t)N_BATCH * P_TOK * C_IN * 2);
    unsigned short* WqT   = (unsigned short*)carve((size_t)C_IN * D_KQ * 2);   // [512][256]
    unsigned short* WkT   = (unsigned short*)carve((size_t)C_IN * D_KQ * 2);   // [512][256]
    unsigned short* WvT   = (unsigned short*)carve((size_t)C_IN * D_V * 2);    // [64][256]
    unsigned short* WpT   = (unsigned short*)carve((size_t)D_CAT * C_IN * 2);  // [256][512]
    unsigned short* Qm    = (unsigned short*)carve((size_t)N_BATCH * P_TOK * D_KQ * 2);
    unsigned short* Km    = (unsigned short*)carve((size_t)N_BATCH * P_TOK * D_KQ * 2);
    unsigned short* Vm    = (unsigned short*)carve((size_t)N_BATCH * P_TOK * D_V * 2);
    unsigned short* Om    = (unsigned short*)carve((size_t)N_BATCH * P_TOK * D_CAT * 2);
    (void)ws_size; (void)in_sizes; (void)n_in; (void)out_size;

    temb_kernel<<<N_BATCH, 256, 0, stream>>>(t, Wt, bt, temb);
    bn_stats_kernel<<<C_IN, 256, 0, stream>>>(x, temb, mean, rstd);
    norm_kernel<<<N_BATCH * P_TOK, 256, 0, stream>>>(x, temb, mean, rstd, gamma, beta, Xbf);

    cvtT_kernel<<<(C_IN * D_KQ + 255) / 256, 256, 0, stream>>>(Wq, WqT, C_IN, D_KQ);
    cvtT_kernel<<<(C_IN * D_KQ + 255) / 256, 256, 0, stream>>>(Wk, WkT, C_IN, D_KQ);
    cvtT_kernel<<<(C_IN * D_V + 255) / 256, 256, 0, stream>>>(Wv, WvT, C_IN, D_V);
    cvtT_kernel<<<(D_CAT * C_IN + 255) / 256, 256, 0, stream>>>(Wp, WpT, D_CAT, C_IN);

    // Q/K/V projections: [1024,256] @ [256,N] per batch (B transposed)
    gemm_bf16_kernel<<<dim3(D_KQ / 64, P_TOK / 16, N_BATCH), 128, 0, stream>>>(
        Xbf, WqT, Qm, C_IN, D_KQ, (long)P_TOK * C_IN, (long)P_TOK * D_KQ);
    gemm_bf16_kernel<<<dim3(D_KQ / 64, P_TOK / 16, N_BATCH), 128, 0, stream>>>(
        Xbf, WkT, Km, C_IN, D_KQ, (long)P_TOK * C_IN, (long)P_TOK * D_KQ);
    gemm_bf16_kernel<<<dim3(D_V / 64, P_TOK / 16, N_BATCH), 128, 0, stream>>>(
        Xbf, WvT, Vm, C_IN, D_V, (long)P_TOK * C_IN, (long)P_TOK * D_V);

    attn_kernel<<<dim3(P_TOK / 16, H_HEADS, N_BATCH), 32, 0, stream>>>(Qm, Km, Vm, Om);

    proj_kernel<<<dim3(C_IN / 64, P_TOK / 16, N_BATCH), 128, 0, stream>>>(Om, WpT, bp, x, out);
}
